// SpatioTemporalGNN_27135603376416
// MI455X (gfx1250) — matrix operand order repaired and checked by reading, not compiled, not run on hardware
//
#include <hip/hip_runtime.h>
#include <hip/hip_bf16.h>

// ---------------------------------------------------------------------------
// SpatioTemporal GNN for MI455X (gfx1250).
// Dense layers run on v_wmma_f32_16x16x32_bf16 (bf16 in, f32 accumulate),
// weight-stationary: each wave caches its 64-col x K weight panel in VGPRs
// and streams 16-row activation tiles over it, so WMMAs issue back-to-back
// after a single A-fragment load wait. Epilogue flags (bias / accumulate)
// are template parameters so the C read-modify-write path is branch-free.
// Graph aggregation (~30 GB edge traffic) is the roofline bottleneck.
// ---------------------------------------------------------------------------

typedef __bf16 bf16_t;
typedef __attribute__((ext_vector_type(16))) __bf16 v16bf;
typedef __attribute__((ext_vector_type(8)))  float  v8f;

// ---------------- elementwise / prep kernels -------------------------------

__global__ void zero_f32_kernel(float* __restrict__ p, long long n) {
    long long i = (long long)blockIdx.x * blockDim.x + threadIdx.x;
    long long stride = (long long)gridDim.x * blockDim.x;
    for (; i < n; i += stride) p[i] = 0.0f;
}

__global__ void cast_f32_bf16_kernel(const float* __restrict__ in, bf16_t* __restrict__ out,
                                     long long n) {
    long long i = (long long)blockIdx.x * blockDim.x + threadIdx.x;
    long long stride = (long long)gridDim.x * blockDim.x;
    for (; i < n; i += stride) out[i] = (bf16_t)in[i];
}

// in[K][Nc] (row-major f32)  ->  out[Nc][K] (row-major bf16)  (B pre-transpose)
__global__ void transpose_cast_kernel(const float* __restrict__ in, bf16_t* __restrict__ out,
                                      int K, int Nc) {
    long long total = (long long)K * Nc;
    long long i = (long long)blockIdx.x * blockDim.x + threadIdx.x;
    long long stride = (long long)gridDim.x * blockDim.x;
    for (; i < total; i += stride) {
        int k = (int)(i / Nc), n = (int)(i % Nc);
        out[(size_t)n * K + k] = (bf16_t)in[i];
    }
}

__global__ void deg_accum_kernel(const int* __restrict__ dst, float* __restrict__ deg, int E) {
    long long i = (long long)blockIdx.x * blockDim.x + threadIdx.x;
    long long stride = (long long)gridDim.x * blockDim.x;
    for (; i < E; i += stride) atomicAdd(&deg[dst[i]], 1.0f);
}

__global__ void deg_finalize_kernel(float* __restrict__ deg, int N) {
    long long i = (long long)blockIdx.x * blockDim.x + threadIdx.x;
    long long stride = (long long)gridDim.x * blockDim.x;
    for (; i < N; i += stride) deg[i] = 1.0f / fmaxf(deg[i], 1.0f);
}

// agg[dst] += x[src]  over E edges, F features (F % 4 == 0)
__global__ void scatter_add_kernel(const float* __restrict__ x, const int* __restrict__ src,
                                   const int* __restrict__ dst, float* __restrict__ agg,
                                   int E, int F) {
    int fq = F >> 2;
    long long total = (long long)E * fq;
    long long i = (long long)blockIdx.x * blockDim.x + threadIdx.x;
    long long stride = (long long)gridDim.x * blockDim.x;
    for (; i < total; i += stride) {
        int e  = (int)(i / fq);
        int f4 = (int)(i % fq) << 2;
        int s = src[e], d = dst[e];
        const float4 v = *(const float4*)(x + (size_t)s * F + f4);
        float* a = agg + (size_t)d * F + f4;
        atomicAdd(a + 0, v.x);
        atomicAdd(a + 1, v.y);
        atomicAdd(a + 2, v.z);
        atomicAdd(a + 3, v.w);
    }
}

// out_bf16[i] = (bf16)(agg[i] * invdeg[row])
__global__ void scale_rows_cast_kernel(const float* __restrict__ agg,
                                       const float* __restrict__ invdeg,
                                       bf16_t* __restrict__ out, int N, int F) {
    long long total = (long long)N * F;
    long long i = (long long)blockIdx.x * blockDim.x + threadIdx.x;
    long long stride = (long long)gridDim.x * blockDim.x;
    for (; i < total; i += stride) {
        int node = (int)(i / F);
        out[i] = (bf16_t)(agg[i] * invdeg[node]);
    }
}

// ---------------- WMMA GEMM (weight stationary) ----------------------------
// C[M x Nc] (+)= A[M x K] * Bt^T  (+ bias)
// A: bf16 row-major.  Bt: bf16 [Nc x K] row-major (i.e. B transposed).
// grid = (m-blocks, Nc/64). Each wave owns one 64-column group: it loads the
// whole 64 x K weight panel into VGPRs once (KSTEPS*4 fragments), then loops
// over 16-row A tiles, batch-loading the tile's A fragments so the WMMA chain
// issues without per-fragment load waits. HAS_BIAS/ACC are compile-time so
// the epilogue is branch-free (C loads clause-batched in the ACC variant).
// Requires: M % 16 == 0, Nc % 64 == 0, K == KSTEPS*32.
template<int KSTEPS, bool HAS_BIAS, bool ACC>
__global__ void __launch_bounds__(128)
wmma_gemm_ws_kernel(const bf16_t* __restrict__ A, const bf16_t* __restrict__ Bt,
                    const float* __restrict__ bias, float* __restrict__ C,
                    int M, int Nc) {
    constexpr int K = KSTEPS * 32;
    const int lane = threadIdx.x & 31;
    const int wIB  = threadIdx.x >> 5;
    const int wavesPerBlock = blockDim.x >> 5;
    const int ng   = blockIdx.y;          // 64-column group
    const int n0g  = ng << 6;
    const int mTiles = M >> 4;

    // ---- B panel -> registers (weight stationary) ----
    // B fragment (32x16 bf16 from transposed Bt): lane l -> column (l&15),
    // K rows kbB..kbB+15 where kbB = (l>>4)*16.
    const int nB  = lane & 15;
    const int kbB = (lane >> 4) << 4;
    v16bf bfrag[KSTEPS][4];
    #pragma unroll
    for (int ks = 0; ks < KSTEPS; ++ks) {
        #pragma unroll
        for (int nt = 0; nt < 4; ++nt) {
            const bf16_t* Bcol = Bt + (size_t)(n0g + (nt << 4) + nB) * K + ks * 32 + kbB;
            #pragma unroll
            for (int j = 0; j < 16; ++j) bfrag[ks][nt][j] = Bcol[j];
        }
    }

    // Per-lane bias values for the 4 n-tiles (uniform across M rows).
    float bv[4] = {0.0f, 0.0f, 0.0f, 0.0f};
    if (HAS_BIAS) {
        #pragma unroll
        for (int nt = 0; nt < 4; ++nt) bv[nt] = bias[n0g + (nt << 4) + (lane & 15)];
    }

    // A fragment (16x32 bf16): lane l -> row (l&15), K rows
    // kbA..kbA+15 where kbA = (l>>4)*16.
    const int kbA  = (lane >> 4) << 4;
    const int mRow = lane & 15;
    const int mStride = gridDim.x * wavesPerBlock;

    for (int mt = blockIdx.x * wavesPerBlock + wIB; mt < mTiles; mt += mStride) {
        const int m0 = mt << 4;
        const bf16_t* Arow = A + (size_t)(m0 + mRow) * K;

        v16bf afrag[KSTEPS];
        #pragma unroll
        for (int ks = 0; ks < KSTEPS; ++ks) {
            #pragma unroll
            for (int j = 0; j < 8; ++j) afrag[ks][j]     = Arow[ks * 32 + kbA + j];
            #pragma unroll
            for (int j = 0; j < 8; ++j) afrag[ks][8 + j] = Arow[ks * 32 + 8 + kbA + j];
        }

        v8f acc[4];
        #pragma unroll
        for (int nt = 0; nt < 4; ++nt) acc[nt] = v8f{};

        #pragma unroll
        for (int ks = 0; ks < KSTEPS; ++ks) {
            #pragma unroll
            for (int nt = 0; nt < 4; ++nt) {
                acc[nt] = __builtin_amdgcn_wmma_f32_16x16x32_bf16(
                    false, afrag[ks], false, bfrag[ks][nt], (short)0, acc[nt], false, false);
            }
        }

        // C/D layout: VGPR r -> M = (lane>>4)*8 + r, N = lane&15.
        const int mBase = m0 + ((lane >> 4) << 3);
        const int nCol  = lane & 15;
        #pragma unroll
        for (int nt = 0; nt < 4; ++nt) {
            const int n = n0g + (nt << 4) + nCol;
            float* cp = C + (size_t)mBase * Nc + n;
            if (ACC) {
                float old[8];
                #pragma unroll
                for (int r = 0; r < 8; ++r) old[r] = cp[(size_t)r * Nc];
                #pragma unroll
                for (int r = 0; r < 8; ++r) {
                    float v = old[r] + acc[nt][r];
                    if (HAS_BIAS) v += bv[nt];
                    cp[(size_t)r * Nc] = v;
                }
            } else {
                #pragma unroll
                for (int r = 0; r < 8; ++r) {
                    float v = acc[nt][r];
                    if (HAS_BIAS) v += bv[nt];
                    cp[(size_t)r * Nc] = v;
                }
            }
        }
    }
}

// ---------------- fused LayerNorm + ReLU (+residual) -----------------------
// One 128-thread block per node (Hdim == 128).
__global__ void __launch_bounds__(128)
ln_relu_kernel(const float* __restrict__ S, const float* __restrict__ g,
               const float* __restrict__ be, const float* __restrict__ res,
               float* __restrict__ outF, bf16_t* __restrict__ outB) {
    __shared__ float s1[128];
    __shared__ float s2[128];
    const int j = threadIdx.x;
    const size_t base = (size_t)blockIdx.x * 128;
    float v = S[base + j];
    s1[j] = v;
    s2[j] = v * v;
    __syncthreads();
    #pragma unroll
    for (int ofs = 64; ofs > 0; ofs >>= 1) {
        if (j < ofs) { s1[j] += s1[j + ofs]; s2[j] += s2[j + ofs]; }
        __syncthreads();
    }
    const float mean = s1[0] * (1.0f / 128.0f);
    const float var  = s2[0] * (1.0f / 128.0f) - mean * mean;
    float y = (v - mean) * rsqrtf(var + 1e-5f) * g[j] + be[j];
    y = fmaxf(y, 0.0f);
    if (res)  y += res[base + j];
    if (outF) outF[base + j] = y;
    outB[base + j] = (bf16_t)y;
}

// ---------------- GRU gate elementwise (torch order r,z,n) -----------------
__global__ void gru_elem_kernel(const float* __restrict__ gi, const float* __restrict__ gh,
                                float* __restrict__ h, bf16_t* __restrict__ hbf,
                                int N, int Hd) {
    long long total = (long long)N * Hd;
    long long i = (long long)blockIdx.x * blockDim.x + threadIdx.x;
    long long stride = (long long)gridDim.x * blockDim.x;
    for (; i < total; i += stride) {
        int node = (int)(i / Hd), j = (int)(i % Hd);
        size_t b = (size_t)node * 3 * Hd;
        float ir = gi[b + j],          hr = gh[b + j];
        float iz = gi[b + Hd + j],     hz = gh[b + Hd + j];
        float in_ = gi[b + 2*Hd + j],  hn = gh[b + 2*Hd + j];
        float r = 1.0f / (1.0f + __expf(-(ir + hr)));
        float z = 1.0f / (1.0f + __expf(-(iz + hz)));
        float ng = tanhf(in_ + r * hn);
        float hv = (1.0f - z) * ng + z * h[i];
        h[i] = hv;
        hbf[i] = (bf16_t)hv;
    }
}

// ---------------- classifier tail: relu(zc) @ W2 + b2 ----------------------
__global__ void cls_final_kernel(const float* __restrict__ zc, const float* __restrict__ W2,
                                 const float* __restrict__ b2, float* __restrict__ out, int N) {
    long long i = (long long)blockIdx.x * blockDim.x + threadIdx.x;
    long long stride = (long long)gridDim.x * blockDim.x;
    for (; i < N; i += stride) {
        const float* zr = zc + (size_t)i * 64;
        float a0 = b2[0], a1 = b2[1];
        #pragma unroll
        for (int j = 0; j < 64; ++j) {
            float v = fmaxf(zr[j], 0.0f);
            a0 += v * W2[j * 2 + 0];
            a1 += v * W2[j * 2 + 1];
        }
        out[i * 2 + 0] = a0;
        out[i * 2 + 1] = a1;
    }
}

// ---------------------------------------------------------------------------

extern "C" void kernel_launch(void* const* d_in, const int* in_sizes, int n_in,
                              void* d_out, int out_size, void* d_ws, size_t ws_size,
                              hipStream_t stream) {
    constexpr int T = 12, F = 64, Hd = 128;
    const int N = out_size / 2;
    const int E = in_sizes[1] / 2;

    const float* x_seq   = (const float*)d_in[0];
    const int*   eidx    = (const int*)  d_in[1];
    const int*   src     = eidx;
    const int*   dst     = eidx + E;
    const float* Wn0     = (const float*)d_in[2];
    const float* Ws0     = (const float*)d_in[3];
    const float* b0      = (const float*)d_in[4];
    const float* g0      = (const float*)d_in[5];
    const float* be0     = (const float*)d_in[6];
    const float* Wn1     = (const float*)d_in[7];
    const float* Ws1     = (const float*)d_in[8];
    const float* b1      = (const float*)d_in[9];
    const float* g1      = (const float*)d_in[10];
    const float* be1     = (const float*)d_in[11];
    const float* gru_Wih = (const float*)d_in[12];
    const float* gru_Whh = (const float*)d_in[13];
    const float* gru_bih = (const float*)d_in[14];
    const float* gru_bhh = (const float*)d_in[15];
    const float* cls_W1  = (const float*)d_in[16];
    const float* cls_b1  = (const float*)d_in[17];
    const float* cls_W2  = (const float*)d_in[18];
    const float* cls_b2  = (const float*)d_in[19];
    float* out = (float*)d_out;

    // ---- workspace layout ----
    char* ws = (char*)d_ws;
    size_t off = 0;
    auto alloc = [&](size_t bytes) -> char* {
        off = (off + 255) & ~(size_t)255;
        char* p = ws + off;
        off += bytes;
        return p;
    };
    float*  invdeg = (float*) alloc((size_t)N * 4);
    bf16_t* xbf    = (bf16_t*)alloc((size_t)N * F * 2);
    float*  agg    = (float*) alloc((size_t)N * Hd * 4);
    bf16_t* aggbf  = (bf16_t*)alloc((size_t)N * Hd * 2);
    float*  sbuf   = (float*) alloc((size_t)N * Hd * 4);
    float*  h0     = (float*) alloc((size_t)N * Hd * 4);
    bf16_t* h0bf   = (bf16_t*)alloc((size_t)N * Hd * 2);
    bf16_t* htbf   = (bf16_t*)alloc((size_t)N * Hd * 2);
    float*  hgru   = (float*) alloc((size_t)N * Hd * 4);
    bf16_t* hbf    = (bf16_t*)alloc((size_t)N * Hd * 2);
    float*  gi     = (float*) alloc((size_t)N * 3 * Hd * 4);
    float*  gh     = (float*) alloc((size_t)N * 3 * Hd * 4);
    float*  zc     = (float*) alloc((size_t)N * 64 * 4);
    bf16_t* Wn0t   = (bf16_t*)alloc((size_t)Hd * F * 2);
    bf16_t* Ws0t   = (bf16_t*)alloc((size_t)Hd * F * 2);
    bf16_t* Wn1t   = (bf16_t*)alloc((size_t)Hd * Hd * 2);
    bf16_t* Ws1t   = (bf16_t*)alloc((size_t)Hd * Hd * 2);
    bf16_t* Wihb   = (bf16_t*)alloc((size_t)3 * Hd * Hd * 2);
    bf16_t* Whhb   = (bf16_t*)alloc((size_t)3 * Hd * Hd * 2);
    bf16_t* W1t    = (bf16_t*)alloc((size_t)64 * Hd * 2);
    (void)ws_size; (void)n_in;

    const dim3 GS(4096), BS(256);

    // Dispatch: KSTEPS in {2,4}, bias-on => ACC off, acc-on => bias off (our use).
    auto gemm = [&](const bf16_t* A, const bf16_t* Bt, const float* bias, float* C,
                    int M, int Nc, int K, int acc) {
        const int wavesPerBlock = 4;  // 128 threads
        int mBlocks = ((M >> 4) + wavesPerBlock - 1) / wavesPerBlock;
        dim3 grid(mBlocks, Nc >> 6);
        if (K == 64) {
            if (acc) wmma_gemm_ws_kernel<2, false, true ><<<grid, 128, 0, stream>>>(A, Bt, bias, C, M, Nc);
            else     wmma_gemm_ws_kernel<2, true,  false><<<grid, 128, 0, stream>>>(A, Bt, bias, C, M, Nc);
        } else {  // K == 128
            if (acc) wmma_gemm_ws_kernel<4, false, true ><<<grid, 128, 0, stream>>>(A, Bt, bias, C, M, Nc);
            else     wmma_gemm_ws_kernel<4, true,  false><<<grid, 128, 0, stream>>>(A, Bt, bias, C, M, Nc);
        }
    };

    // ---- one-time prep (recomputed every launch; deterministic) ----
    zero_f32_kernel<<<GS, BS, 0, stream>>>(invdeg, N);
    deg_accum_kernel<<<GS, BS, 0, stream>>>(dst, invdeg, E);
    deg_finalize_kernel<<<GS, BS, 0, stream>>>(invdeg, N);

    transpose_cast_kernel<<<GS, BS, 0, stream>>>(Wn0, Wn0t, F, Hd);
    transpose_cast_kernel<<<GS, BS, 0, stream>>>(Ws0, Ws0t, F, Hd);
    transpose_cast_kernel<<<GS, BS, 0, stream>>>(Wn1, Wn1t, Hd, Hd);
    transpose_cast_kernel<<<GS, BS, 0, stream>>>(Ws1, Ws1t, Hd, Hd);
    transpose_cast_kernel<<<GS, BS, 0, stream>>>(cls_W1, W1t, Hd, 64);
    // x @ Wih.T : B = Wih.T, so Bt = Wih as-is (straight cast); same for Whh.
    cast_f32_bf16_kernel<<<GS, BS, 0, stream>>>(gru_Wih, Wihb, (long long)3 * Hd * Hd);
    cast_f32_bf16_kernel<<<GS, BS, 0, stream>>>(gru_Whh, Whhb, (long long)3 * Hd * Hd);

    // GRU state := 0 (f32 + bf16 mirror)
    zero_f32_kernel<<<GS, BS, 0, stream>>>(hgru, (long long)N * Hd);
    zero_f32_kernel<<<GS, BS, 0, stream>>>((float*)hbf, (long long)N * Hd / 2);

    // ---- temporal loop: spatial block then fused GRU step -----------------
    for (int t = 0; t < T; ++t) {
        const float* x_t = x_seq + (size_t)t * N * F;
        cast_f32_bf16_kernel<<<GS, BS, 0, stream>>>(x_t, xbf, (long long)N * F);

        // SAGE layer 0
        zero_f32_kernel<<<GS, BS, 0, stream>>>(agg, (long long)N * F);
        scatter_add_kernel<<<GS, BS, 0, stream>>>(x_t, src, dst, agg, E, F);
        scale_rows_cast_kernel<<<GS, BS, 0, stream>>>(agg, invdeg, aggbf, N, F);
        gemm(aggbf, Wn0t, b0, sbuf, N, Hd, F, 0);        // s  = agg @ Wn0 + b0
        gemm(xbf,   Ws0t, nullptr, sbuf, N, Hd, F, 1);   // s += x   @ Ws0
        ln_relu_kernel<<<N, 128, 0, stream>>>(sbuf, g0, be0, nullptr, h0, h0bf);

        // SAGE layer 1 (+ residual)
        zero_f32_kernel<<<GS, BS, 0, stream>>>(agg, (long long)N * Hd);
        scatter_add_kernel<<<GS, BS, 0, stream>>>(h0, src, dst, agg, E, Hd);
        scale_rows_cast_kernel<<<GS, BS, 0, stream>>>(agg, invdeg, aggbf, N, Hd);
        gemm(aggbf, Wn1t, b1, sbuf, N, Hd, Hd, 0);       // s  = agg @ Wn1 + b1
        gemm(h0bf,  Ws1t, nullptr, sbuf, N, Hd, Hd, 1);  // s += h0  @ Ws1
        ln_relu_kernel<<<N, 128, 0, stream>>>(sbuf, g1, be1, h0, nullptr, htbf);

        // GRU step
        gemm(htbf, Wihb, gru_bih, gi, N, 3 * Hd, Hd, 0); // gi = ht @ Wih.T + bih
        gemm(hbf,  Whhb, gru_bhh, gh, N, 3 * Hd, Hd, 0); // gh = h  @ Whh.T + bhh
        gru_elem_kernel<<<GS, BS, 0, stream>>>(gi, gh, hgru, hbf, N, Hd);
    }

    // ---- classifier -------------------------------------------------------
    gemm(hbf, W1t, cls_b1, zc, N, 64, Hd, 0);            // zc = hT @ W1 + b1
    cls_final_kernel<<<(N + 255) / 256, 256, 0, stream>>>(zc, cls_W2, cls_b2, out, N);
}
